// SelectiveStateSpaceModel_54494545051920
// MI455X (gfx1250) — matrix operand order repaired and checked by reading, not compiled, not run on hardware
//
#include <hip/hip_runtime.h>
#include <hip/hip_bf16.h>

// ---------------------------------------------------------------------------
// Selective SSM (Mamba-style) for MI455X / gfx1250, wave32 + WMMA bf16 + TDM.
//
//   1) cvt f32->bf16: x, W_dt, W_B, W_C
//   2) fused dual GEMM (TDM double-buffered LDS staging):
//        dt_raw = x*W_dt^T, Bp = x*W_B^T   (v_wmma_f32_16x16x32_bf16)
//      epilogue: dt=softplus(+b), abar=exp(-dt*exp(A_log)), inp=dt*Bp*x
//   3) chunked parallel scan over S (associative op), fused bf16 cvt of h
//   4) GEMM: out = h*W_C^T + D_skip*x   (TDM staged, WMMA)
// ---------------------------------------------------------------------------

typedef unsigned short u16;
typedef __attribute__((ext_vector_type(16))) __bf16 v16bf;
typedef __attribute__((ext_vector_type(8)))  float  v8f;
typedef __attribute__((ext_vector_type(4)))  unsigned int u32x4;
typedef __attribute__((ext_vector_type(8)))  int i32x8;
typedef __attribute__((ext_vector_type(4)))  int i32x4;

#define BATCH   4
#define SEQ     2048
#define DIM     2048              // D == N
#define MROWS   (BATCH * SEQ)     // 8192
#define KDIM    DIM
#define NCHUNK  16
#define CLEN    (SEQ / NCHUNK)    // 128

// K-tile of 64 bf16 per stage (2 WMMA k-substeps). TDM pads each 128B row
// with 16B -> LDS row stride 144B = 72 halves (bank-conflict friendly).
#define BK        64
#define LDSTR     72              // halves
#define KT_STEPS  (KDIM / BK)     // 32

#if defined(__has_builtin)
#  if __has_builtin(__builtin_amdgcn_tensor_load_to_lds) && \
      __has_builtin(__builtin_amdgcn_s_wait_tensorcnt)
#    define HAVE_TDM 1
#  endif
#endif
#ifndef HAVE_TDM
#  define HAVE_TDM 0
#endif

// ---------------------------------------------------------------------------
__device__ __forceinline__ u16 f2bf(float f) {
    unsigned u = __float_as_uint(f);
    unsigned r = u + 0x7FFFu + ((u >> 16) & 1u);   // round-to-nearest-even
    return (u16)(r >> 16);
}

__device__ __forceinline__ v8f wmma_bf16(v16bf a, v16bf b, v8f c) {
    return __builtin_amdgcn_wmma_f32_16x16x32_bf16(false, a, false, b,
                                                   (short)0, c, false, false);
}

// ---------------------------------------------------------------------------
// TDM: 2-D bf16 tile load, global -> LDS, with per-row LDS padding.
//   tile_d0 elements contiguous (k), tile_d1 rows; tensor row stride = str_d0.
//   Row = tile_d0*2 bytes; pad 16B every 128B  (pad_interval=4 -> 32 DWORDs,
//   pad_amount=3 -> 4 DWORDs). Requires tile_d0 == 64.
__device__ __forceinline__ void tdm_load_2d(unsigned lds_off, const u16* gptr,
                                            unsigned tile_d0, unsigned tile_d1,
                                            unsigned tensor_d0, unsigned tensor_d1,
                                            unsigned str_d0) {
#if HAVE_TDM
    unsigned long long ga = (unsigned long long)(const void*)gptr;
    u32x4 g0;
    g0.x = 1u;                                         // count=1, user mode
    g0.y = lds_off;                                    // LDS byte address
    g0.z = (unsigned)(ga & 0xFFFFFFFFu);               // global_addr[31:0]
    g0.w = (unsigned)((ga >> 32) & 0x01FFFFFFu) | (2u << 30);  // [56:32]|type=2
    i32x8 g1;
    g1[0] = (int)((1u << 16)        // data_size = 2 bytes
                | (1u << 20)        // pad_enable
                | (4u << 22)        // pad_interval: 32 DWORDs (128B)
                | (3u << 25));      // pad_amount:   4 DWORDs  (16B)
    g1[1] = (int)((tensor_d0 & 0xFFFFu) << 16);                         // dim0 lo
    g1[2] = (int)(((tensor_d0 >> 16) & 0xFFFFu) | ((tensor_d1 & 0xFFFFu) << 16));
    g1[3] = (int)(((tensor_d1 >> 16) & 0xFFFFu) | ((tile_d0 & 0xFFFFu) << 16));
    g1[4] = (int)(tile_d1 & 0xFFFFu);                                   // tile_dim1
    g1[5] = (int)str_d0;                                                // stride lo32
    g1[6] = 0;
    g1[7] = 0;
    i32x4 z4 = {0, 0, 0, 0};
#if __clang_major__ >= 23
    i32x8 z8 = {0, 0, 0, 0, 0, 0, 0, 0};
    __builtin_amdgcn_tensor_load_to_lds(g0, g1, z4, z4, z8, 0);
#else
    __builtin_amdgcn_tensor_load_to_lds(g0, g1, z4, z4, 0);
#endif
#else
    (void)lds_off; (void)gptr; (void)tile_d0; (void)tile_d1;
    (void)tensor_d0; (void)tensor_d1; (void)str_d0;
#endif
}

__device__ __forceinline__ void tdm_wait() {
#if HAVE_TDM
    __builtin_amdgcn_s_wait_tensorcnt(0);
#endif
}

__device__ __forceinline__ unsigned lds_byte_off(const void* p) {
    return (unsigned)(unsigned long long)(uintptr_t)p;
}

#if !HAVE_TDM
// Fallback staging (round-1 style) if TDM builtin is unavailable.
__device__ __forceinline__ void stage_tile(const u16* __restrict__ g, u16* s,
                                           int rows, int tid, int nthreads) {
    // rows x 64 halves, LDS stride LDSTR
    int total = rows * 8;                  // 16B segments
    for (int lin = tid; lin < total; lin += nthreads) {
        int row = lin >> 3, seg = lin & 7;
        *(uint4*)&s[row * LDSTR + seg * 8] =
            *(const uint4*)(g + (size_t)row * KDIM + seg * 8);
    }
}
#endif

// ---------------------------------------------------------------------------
// A fragment 16x32 (MxK): lanes 0-15 -> rows m0.., K {0..7,16..23};
// lanes 16-31 -> same rows, K {8..15,24..31}.  ks = k-substep offset (halves).
__device__ __forceinline__ v16bf load_frag_a(const u16* __restrict__ s,
                                             int m0, int ks, int lane) {
    int row = m0 + (lane & 15);
    int c0  = (lane >> 4) << 3;
    union { uint4 q[2]; v16bf v; } u;
    const u16* p = s + row * LDSTR + ks + c0;
    u.q[0] = *(const uint4*)(p);
    u.q[1] = *(const uint4*)(p + 16);
    return u.v;
}

// B fragment 32x16 (KxN): lane n holds col n; lanes 0-15 K {0..7,16..23}, 16-31 K {8..15,24..31}.
__device__ __forceinline__ v16bf load_frag_b(const u16* __restrict__ s,
                                             int n0, int ks, int lane) {
    int row = n0 + (lane & 15);
    int k0  = (lane >> 4) << 3;
    union { uint4 q[2]; v16bf v; } u;
    const u16* p = s + row * LDSTR + ks + k0;
    u.q[0] = *(const uint4*)(p);
    u.q[1] = *(const uint4*)(p + 16);
    return u.v;
}

// ---------------------------------------------------------------------------
__global__ void cvt_bf16_kernel(const float* __restrict__ src,
                                u16* __restrict__ dst, int n) {
    int i = blockIdx.x * 256 + threadIdx.x;
    if (i < n) dst[i] = f2bf(src[i]);
}

// ---------------------------------------------------------------------------
// Fused dual GEMM + SSM epilogue.  Block tile 128(M) x 64(N), K-stage 64,
// 8 waves (4x2), 32x32 per wave -> 16 WMMA per wave per stage per weight pair.
__global__ __launch_bounds__(256)
void gemm_dual_ssm_kernel(const u16*  __restrict__ xbf,
                          const u16*  __restrict__ wdt,
                          const u16*  __restrict__ wb,
                          const float* __restrict__ b_dt,
                          const float* __restrict__ a_log,
                          const float* __restrict__ xf,
                          float* __restrict__ abar,
                          float* __restrict__ inp) {
    const int mBase = blockIdx.y * 128;
    const int nBase = blockIdx.x * 64;
    const int tid   = threadIdx.x;
    const int lane  = tid & 31;
    const int wv    = tid >> 5;
    const int waveM = (wv & 3) * 32;
    const int waveN = (wv >> 2) * 32;

    __shared__ __align__(16) u16 As[2][128 * LDSTR];
    __shared__ __align__(16) u16 Bs[2][2][64 * LDSTR];

    v8f accDt[2][2] = {};
    v8f accBp[2][2] = {};

    const u16* gA  = xbf + (size_t)mBase * KDIM;
    const u16* gB0 = wdt + (size_t)nBase * KDIM;
    const u16* gB1 = wb  + (size_t)nBase * KDIM;

#if HAVE_TDM
    if (wv == 0) {   // TDM: EXEC-independent, per-wave; wave 0 drives the DMA
        tdm_load_2d(lds_byte_off(&As[0][0]),    gA,  BK, 128, KDIM, MROWS, KDIM);
        tdm_load_2d(lds_byte_off(&Bs[0][0][0]), gB0, BK, 64,  KDIM, DIM,   KDIM);
        tdm_load_2d(lds_byte_off(&Bs[0][1][0]), gB1, BK, 64,  KDIM, DIM,   KDIM);
        tdm_wait();
    }
    __syncthreads();
#else
    stage_tile(gA,  As[0],    128, tid, 256);
    stage_tile(gB0, Bs[0][0], 64,  tid, 256);
    stage_tile(gB1, Bs[0][1], 64,  tid, 256);
    __syncthreads();
#endif

    for (int kt = 0; kt < KT_STEPS; ++kt) {
        const int cur = kt & 1, nxt = cur ^ 1;
        const int kOff = (kt + 1) * BK;
#if HAVE_TDM
        if (wv == 0 && kt + 1 < KT_STEPS) {   // prefetch next stage via TDM
            tdm_load_2d(lds_byte_off(&As[nxt][0]),    gA + kOff,  BK, 128, KDIM, MROWS, KDIM);
            tdm_load_2d(lds_byte_off(&Bs[nxt][0][0]), gB0 + kOff, BK, 64,  KDIM, DIM,   KDIM);
            tdm_load_2d(lds_byte_off(&Bs[nxt][1][0]), gB1 + kOff, BK, 64,  KDIM, DIM,   KDIM);
        }
#endif
#pragma unroll
        for (int s = 0; s < 2; ++s) {         // two 32-wide k-substeps
            const int ks = s * 32;
            v16bf af0 = load_frag_a(As[cur], waveM,      ks, lane);
            v16bf af1 = load_frag_a(As[cur], waveM + 16, ks, lane);
            v16bf bd0 = load_frag_b(Bs[cur][0], waveN,      ks, lane);
            v16bf bd1 = load_frag_b(Bs[cur][0], waveN + 16, ks, lane);
            v16bf bb0 = load_frag_b(Bs[cur][1], waveN,      ks, lane);
            v16bf bb1 = load_frag_b(Bs[cur][1], waveN + 16, ks, lane);

            accDt[0][0] = wmma_bf16(af0, bd0, accDt[0][0]);
            accDt[0][1] = wmma_bf16(af0, bd1, accDt[0][1]);
            accDt[1][0] = wmma_bf16(af1, bd0, accDt[1][0]);
            accDt[1][1] = wmma_bf16(af1, bd1, accDt[1][1]);
            accBp[0][0] = wmma_bf16(af0, bb0, accBp[0][0]);
            accBp[0][1] = wmma_bf16(af0, bb1, accBp[0][1]);
            accBp[1][0] = wmma_bf16(af1, bb0, accBp[1][0]);
            accBp[1][1] = wmma_bf16(af1, bb1, accBp[1][1]);
        }
#if HAVE_TDM
        if (wv == 0) tdm_wait();              // next buffers complete
        __syncthreads();
#else
        __syncthreads();
        if (kt + 1 < KT_STEPS) {
            stage_tile(gA + kOff,  As[nxt],    128, tid, 256);
            stage_tile(gB0 + kOff, Bs[nxt][0], 64,  tid, 256);
            stage_tile(gB1 + kOff, Bs[nxt][1], 64,  tid, 256);
        }
        __syncthreads();
#endif
    }

    // epilogue: dt=softplus(+b), abar=exp(-dt*exp(A_log)), inp=dt*Bp*x
#pragma unroll
    for (int f = 0; f < 2; ++f) {
#pragma unroll
        for (int g = 0; g < 2; ++g) {
            int n  = nBase + waveN + g * 16 + (lane & 15);
            int m0 = mBase + waveM + f * 16 + ((lane >> 4) << 3);
            float bias = b_dt[n];
            float aexp = __expf(a_log[n]);
            v8f cdt = accDt[f][g];
            v8f cbp = accBp[f][g];
#pragma unroll
            for (int i = 0; i < 8; ++i) {
                size_t idx = (size_t)(m0 + i) * DIM + n;
                float z   = cdt[i] + bias;
                float dtv = (z > 20.f) ? z : log1pf(__expf(z));   // softplus
                float ab  = __expf(-dtv * aexp);
                float u   = dtv * cbp[i] * xf[idx];
                abar[idx] = ab;
                inp[idx]  = u;
            }
        }
    }
}

// ---------------------------------------------------------------------------
// Scan phase 1: per (b, chunk, n) local scan; store h_local and chunk A-prod.
__global__ void scan_phase1_kernel(const float* __restrict__ abar,
                                   const float* __restrict__ inp,
                                   float* __restrict__ hloc,
                                   float* __restrict__ aprod) {
    const int nb = DIM / 256;
    int bid = blockIdx.x;
    int n   = (bid % nb) * 256 + threadIdx.x;
    int rest = bid / nb;
    int c = rest % NCHUNK;
    int b = rest / NCHUNK;

    float h = 0.f, p = 1.f;
    int t0 = c * CLEN;
#pragma unroll 4
    for (int t = t0; t < t0 + CLEN; ++t) {
        size_t idx = (size_t)(b * SEQ + t) * DIM + n;
        float a = abar[idx];
        h = a * h + inp[idx];
        p *= a;
        hloc[idx] = h;
    }
    aprod[(size_t)(b * NCHUNK + c) * DIM + n] = p;
}

// Scan phase 2: sequential scan over the 16 chunk aggregates.
__global__ void scan_phase2_kernel(const float* __restrict__ hloc,
                                   const float* __restrict__ aprod,
                                   float* __restrict__ carry) {
    int ch = blockIdx.x * 256 + threadIdx.x;   // 0..B*DIM-1
    int b = ch / DIM, n = ch % DIM;
    float cin = 0.f;
#pragma unroll
    for (int c = 0; c < NCHUNK; ++c) {
        size_t ci = (size_t)(b * NCHUNK + c) * DIM + n;
        carry[ci] = cin;
        float hend = hloc[(size_t)(b * SEQ + c * CLEN + CLEN - 1) * DIM + n];
        cin = hend + aprod[ci] * cin;
    }
}

// Scan phase 3: h = h_local + prefixprod(a)*carry, fused bf16 conversion.
__global__ void scan_phase3_kernel(const float* __restrict__ abar,
                                   const float* __restrict__ hloc,
                                   const float* __restrict__ carry,
                                   u16* __restrict__ hbf) {
    const int nb = DIM / 256;
    int bid = blockIdx.x;
    int n   = (bid % nb) * 256 + threadIdx.x;
    int rest = bid / nb;
    int c = rest % NCHUNK;
    int b = rest / NCHUNK;

    float cin = carry[(size_t)(b * NCHUNK + c) * DIM + n];
    float p = 1.f;
    int t0 = c * CLEN;
#pragma unroll 4
    for (int t = t0; t < t0 + CLEN; ++t) {
        size_t idx = (size_t)(b * SEQ + t) * DIM + n;
        p *= abar[idx];
        float h = hloc[idx] + p * cin;
        hbf[idx] = f2bf(h);
    }
}

// ---------------------------------------------------------------------------
// Output GEMM: out = h*W_C^T + D_skip*x.  Same TDM-staged tiling.
__global__ __launch_bounds__(256)
void gemm_out_kernel(const u16*  __restrict__ hbf,
                     const u16*  __restrict__ wc,
                     const float* __restrict__ dskip,
                     const float* __restrict__ xf,
                     float* __restrict__ out) {
    const int mBase = blockIdx.y * 128;
    const int nBase = blockIdx.x * 64;
    const int tid   = threadIdx.x;
    const int lane  = tid & 31;
    const int wv    = tid >> 5;
    const int waveM = (wv & 3) * 32;
    const int waveN = (wv >> 2) * 32;

    __shared__ __align__(16) u16 As[2][128 * LDSTR];
    __shared__ __align__(16) u16 Bs[2][64 * LDSTR];

    v8f acc[2][2] = {};

    const u16* gA = hbf + (size_t)mBase * KDIM;
    const u16* gB = wc  + (size_t)nBase * KDIM;

#if HAVE_TDM
    if (wv == 0) {
        tdm_load_2d(lds_byte_off(&As[0][0]), gA, BK, 128, KDIM, MROWS, KDIM);
        tdm_load_2d(lds_byte_off(&Bs[0][0]), gB, BK, 64,  KDIM, DIM,   KDIM);
        tdm_wait();
    }
    __syncthreads();
#else
    stage_tile(gA, As[0], 128, tid, 256);
    stage_tile(gB, Bs[0], 64,  tid, 256);
    __syncthreads();
#endif

    for (int kt = 0; kt < KT_STEPS; ++kt) {
        const int cur = kt & 1, nxt = cur ^ 1;
        const int kOff = (kt + 1) * BK;
#if HAVE_TDM
        if (wv == 0 && kt + 1 < KT_STEPS) {
            tdm_load_2d(lds_byte_off(&As[nxt][0]), gA + kOff, BK, 128, KDIM, MROWS, KDIM);
            tdm_load_2d(lds_byte_off(&Bs[nxt][0]), gB + kOff, BK, 64,  KDIM, DIM,   KDIM);
        }
#endif
#pragma unroll
        for (int s = 0; s < 2; ++s) {
            const int ks = s * 32;
            v16bf af0 = load_frag_a(As[cur], waveM,      ks, lane);
            v16bf af1 = load_frag_a(As[cur], waveM + 16, ks, lane);
            v16bf b0  = load_frag_b(Bs[cur], waveN,      ks, lane);
            v16bf b1  = load_frag_b(Bs[cur], waveN + 16, ks, lane);

            acc[0][0] = wmma_bf16(af0, b0, acc[0][0]);
            acc[0][1] = wmma_bf16(af0, b1, acc[0][1]);
            acc[1][0] = wmma_bf16(af1, b0, acc[1][0]);
            acc[1][1] = wmma_bf16(af1, b1, acc[1][1]);
        }
#if HAVE_TDM
        if (wv == 0) tdm_wait();
        __syncthreads();
#else
        __syncthreads();
        if (kt + 1 < KT_STEPS) {
            stage_tile(gA + kOff, As[nxt], 128, tid, 256);
            stage_tile(gB + kOff, Bs[nxt], 64,  tid, 256);
        }
        __syncthreads();
#endif
    }

#pragma unroll
    for (int f = 0; f < 2; ++f) {
#pragma unroll
        for (int g = 0; g < 2; ++g) {
            int d  = nBase + waveN + g * 16 + (lane & 15);
            int m0 = mBase + waveM + f * 16 + ((lane >> 4) << 3);
            float ds = dskip[d];
            v8f c = acc[f][g];
#pragma unroll
            for (int i = 0; i < 8; ++i) {
                size_t idx = (size_t)(m0 + i) * DIM + d;
                out[idx] = c[i] + ds * xf[idx];
            }
        }
    }
}

// ---------------------------------------------------------------------------
extern "C" void kernel_launch(void* const* d_in, const int* in_sizes, int n_in,
                              void* d_out, int out_size, void* d_ws, size_t ws_size,
                              hipStream_t stream) {
    const float* x      = (const float*)d_in[0];   // [B,S,D]
    const float* W_dt   = (const float*)d_in[1];   // [N,D]
    const float* b_dt   = (const float*)d_in[2];   // [N]
    const float* W_B    = (const float*)d_in[3];   // [N,D]
    const float* W_C    = (const float*)d_in[4];   // [D,N]
    const float* A_log  = (const float*)d_in[5];   // [N]
    const float* D_skip = (const float*)d_in[6];   // [D]
    float* out = (float*)d_out;

    char* ws = (char*)d_ws;
    size_t off = 0;
    auto alloc = [&](size_t bytes) {
        size_t o = off;
        off = (off + bytes + 255) & ~(size_t)255;
        return ws + o;
    };
    const size_t MN = (size_t)MROWS * DIM;
    const size_t WN = (size_t)DIM * DIM;

    u16*   x_bf   = (u16*)  alloc(MN * 2);
    u16*   wdt_bf = (u16*)  alloc(WN * 2);
    u16*   wb_bf  = (u16*)  alloc(WN * 2);
    u16*   wc_bf  = (u16*)  alloc(WN * 2);
    float* abar   = (float*)alloc(MN * 4);
    float* inp    = (float*)alloc(MN * 4);
    float* hloc   = (float*)alloc(MN * 4);
    u16*   h_bf   = (u16*)  alloc(MN * 2);
    float* aprod  = (float*)alloc((size_t)BATCH * NCHUNK * DIM * 4);
    float* carry  = (float*)alloc((size_t)BATCH * NCHUNK * DIM * 4);
    (void)ws_size; (void)n_in; (void)in_sizes; (void)out_size;

    cvt_bf16_kernel<<<(int)((MN + 255) / 256), 256, 0, stream>>>(x,    x_bf,   (int)MN);
    cvt_bf16_kernel<<<(int)((WN + 255) / 256), 256, 0, stream>>>(W_dt, wdt_bf, (int)WN);
    cvt_bf16_kernel<<<(int)((WN + 255) / 256), 256, 0, stream>>>(W_B,  wb_bf,  (int)WN);
    cvt_bf16_kernel<<<(int)((WN + 255) / 256), 256, 0, stream>>>(W_C,  wc_bf,  (int)WN);

    dim3 gGemm(DIM / 64, MROWS / 128);       // (32, 64)
    gemm_dual_ssm_kernel<<<gGemm, 256, 0, stream>>>(
        x_bf, wdt_bf, wb_bf, b_dt, A_log, x, abar, inp);

    int scanBlocks = BATCH * NCHUNK * (DIM / 256);   // 512
    scan_phase1_kernel<<<scanBlocks, 256, 0, stream>>>(abar, inp, hloc, aprod);
    scan_phase2_kernel<<<(BATCH * DIM) / 256, 256, 0, stream>>>(hloc, aprod, carry);
    scan_phase3_kernel<<<scanBlocks, 256, 0, stream>>>(abar, hloc, carry, h_bf);

    gemm_out_kernel<<<gGemm, 256, 0, stream>>>(h_bf, wc_bf, D_skip, x, out);
}
